// CrossAttention_56659208569148
// MI455X (gfx1250) — matrix operand order repaired and checked by reading, not compiled, not run on hardware
//
#include <hip/hip_runtime.h>

// ---------------------------------------------------------------------------
// CrossAttention for MI455X (gfx1250, wave32, WMMA + Tensor Data Mover).
// Stage 1: q/k/v 1x1-conv projections via v_wmma_f32_16x16x32_bf16.
// Stage 2: flash-attention (online softmax), both GEMMs on WMMA bf16,
//          K/V tiles staged LDS-side by TDM (tensor_load_to_lds) with
//          double-buffering on TENSORcnt.
// Workspace: qT[b][n][c], kT[b][n][c], v[b][c][n] in bf16 (48 MB).
// ---------------------------------------------------------------------------

#define DEVFN __device__ __forceinline__

typedef __attribute__((ext_vector_type(16))) __bf16 v16bf;
typedef __attribute__((ext_vector_type(8)))  __bf16 v8bf;
typedef __attribute__((ext_vector_type(8)))  float  v8f;
typedef __attribute__((ext_vector_type(4)))  float  v4f;
typedef __attribute__((ext_vector_type(4)))  unsigned int u32x4;
typedef __attribute__((ext_vector_type(4)))  int          i32x4;
typedef __attribute__((ext_vector_type(8)))  int          i32x8;

constexpr int BATCH = 8;
constexpr int CH    = 256;
constexpr int NPIX  = 4096;

DEVFN v16bf cat16(v8bf a, v8bf b) {
  return __builtin_shufflevector(a, b, 0,1,2,3,4,5,6,7,8,9,10,11,12,13,14,15);
}
// 16 contiguous bf16 (operand order k = 16*hi + s used here): two b128 loads.
DEVFN v16bf load16(const __bf16* p) {
  return cat16(*(const v8bf*)p, *(const v8bf*)(p + 8));
}
// WMMA A-operand per-lane pattern: 8 halves at +0 and 8 halves at +16.
DEVFN v16bf loadA(const __bf16* p) {
  return cat16(*(const v8bf*)p, *(const v8bf*)(p + 16));
}
DEVFN v8f wmma_bf16(v16bf a, v16bf b, v8f c) {
  // (neg_a, A, neg_b, B, c_mod, C, reuse_a, reuse_b)
  return __builtin_amdgcn_wmma_f32_16x16x32_bf16(false, a, false, b, (short)0, c,
                                                 false, false);
}
// A-operand tile of a row-major f32 weight matrix W[d][c] (256 cols).
DEVFN v16bf load_w(const float* __restrict__ W, int d, int c0, int hi) {
  const float* p = W + (size_t)d * CH + c0 + hi * 8;
  v4f a0 = *(const v4f*)(p);
  v4f a1 = *(const v4f*)(p + 4);
  v4f b0 = *(const v4f*)(p + 16);
  v4f b1 = *(const v4f*)(p + 20);
  v16bf r;
#pragma unroll
  for (int e = 0; e < 4; ++e) {
    r[e]      = (__bf16)a0[e];
    r[e + 4]  = (__bf16)a1[e];
    r[e + 8]  = (__bf16)b0[e];
    r[e + 12] = (__bf16)b1[e];
  }
  return r;
}

// ---------------------------------------------------------------------------
// Tensor Data Mover: 2D tile (bf16 elements) global -> LDS, with optional
// LDS row padding expressed in the descriptor (D#, ISA 8.3/8.4).
//   addr = global_addr + 2 * (x + y * stride_elems),  x < tile_x, y < tile_y
//   LDS gets pad_amount(code+1) dwords inserted every 2^(pad_code+1) dwords.
// ---------------------------------------------------------------------------
DEVFN void tdm_load_2d(unsigned lds_addr, const void* gaddr,
                       unsigned tile_x, unsigned tile_y,
                       unsigned tensor_x, unsigned tensor_y,
                       unsigned stride_elems,
                       unsigned pad_interval_code, unsigned pad_amount_code)
{
  unsigned long long ga = (unsigned long long)gaddr;
  u32x4 g0;
  g0[0] = 1u;                                          // count=1, user mode
  g0[1] = lds_addr;                                    // LDS byte address
  g0[2] = (unsigned)(ga & 0xFFFFFFFFu);                // global_addr[31:0]
  g0[3] = (unsigned)((ga >> 32) & 0x01FFFFFFu)         // global_addr[56:32]
        | 0x80000000u;                                 // type = 2 ("image")
  i32x8 g1;
  g1[0] = (int)((1u << 16)                             // data_size = 2 bytes
              | (1u << 20)                             // pad_enable
              | (pad_interval_code << 22)
              | (pad_amount_code << 25));
  g1[1] = (int)((tensor_x & 0xFFFFu) << 16);           // tensor_dim0[15:0]
  g1[2] = (int)(((tensor_x >> 16) & 0xFFFFu)           // tensor_dim0[31:16]
              | ((tensor_y & 0xFFFFu) << 16));         // tensor_dim1[15:0]
  g1[3] = (int)(((tensor_y >> 16) & 0xFFFFu)           // tensor_dim1[31:16]
              | ((tile_x & 0xFFFFu) << 16));           // tile_dim0
  g1[4] = (int)(tile_y & 0xFFFFu);                     // tile_dim1 (tile_dim2=0)
  g1[5] = (int)stride_elems;                           // tensor_dim0_stride[31:0]
  g1[6] = 0;                                           // stride hi / dim1_stride
  g1[7] = 0;
  i32x4 z4 = {0, 0, 0, 0};
#if defined(__clang_major__) && (__clang_major__ >= 23)
  i32x8 z8 = {0, 0, 0, 0, 0, 0, 0, 0};
  __builtin_amdgcn_tensor_load_to_lds(g0, g1, z4, z4, z8, 0);
#else
  __builtin_amdgcn_tensor_load_to_lds(g0, g1, z4, z4, 0);
#endif
}

// ---------------------------------------------------------------------------
// Stage 1: qT/kT/v projections. Grid: (NPIX/32, BATCH), 256 threads (8 waves).
// ---------------------------------------------------------------------------
__global__ __launch_bounds__(256, 1) void qkv_project_kernel(
    const float* __restrict__ rgb, const float* __restrict__ hsv,
    const float* __restrict__ Wq, const float* __restrict__ bq,
    const float* __restrict__ Wk, const float* __restrict__ bk,
    const float* __restrict__ Wv, const float* __restrict__ bv,
    __bf16* __restrict__ qT, __bf16* __restrict__ kT, __bf16* __restrict__ vW)
{
  constexpr int TT = 32;    // pixel tile per block
  constexpr int XP = 264;   // padded LDS pitch
  __shared__ __bf16 xh[TT][XP];   // hsv tile, [t][c]
  __shared__ __bf16 xr[TT][XP];   // rgb tile, [t][c]

  const int b   = blockIdx.y;
  const int t0  = blockIdx.x * TT;
  const int tid = threadIdx.x;
  const int w = tid >> 5, lane = tid & 31;
  const int lane16 = lane & 15, hi = lane >> 4;

  // Stage feature tiles (coalesced row read, transposed LDS write).
  {
    const int c = tid;  // 256 threads <-> 256 channels
    const float* hrow = hsv + ((size_t)b * CH + c) * NPIX + t0;
    const float* rrow = rgb + ((size_t)b * CH + c) * NPIX + t0;
#pragma unroll
    for (int t = 0; t < TT; t += 4) {
      v4f hv = *(const v4f*)(hrow + t);
      v4f rv = *(const v4f*)(rrow + t);
#pragma unroll
      for (int u = 0; u < 4; ++u) {
        xh[t + u][c] = (__bf16)hv[u];
        xr[t + u][c] = (__bf16)rv[u];
      }
    }
  }
  __syncthreads();

#pragma unroll
  for (int dd = 0; dd < 2; ++dd) {
    const int dt = w * 2 + dd;          // 8 waves x 2 = 16 d-tiles
    const int d0 = dt * 16;
    const v4f bqv0 = *(const v4f*)(bq + d0 + hi * 8);
    const v4f bqv1 = *(const v4f*)(bq + d0 + hi * 8 + 4);
    const v4f bkv0 = *(const v4f*)(bk + d0 + hi * 8);
    const v4f bkv1 = *(const v4f*)(bk + d0 + hi * 8 + 4);
    const v4f bvv0 = *(const v4f*)(bv + d0 + hi * 8);
    const v4f bvv1 = *(const v4f*)(bv + d0 + hi * 8 + 4);
#pragma unroll
    for (int nt = 0; nt < 2; ++nt) {
      v8f aq = {0.f,0.f,0.f,0.f,0.f,0.f,0.f,0.f};
      v8f ak = aq, av = aq;
#pragma unroll
      for (int kc = 0; kc < 8; ++kc) {      // K = 256 in chunks of 32
        const int c0 = kc * 32;
        v16bf bh = loadA(&xh[nt * 16 + lane16][c0 + hi * 8]);
        v16bf br = loadA(&xr[nt * 16 + lane16][c0 + hi * 8]);
        v16bf wq_t = load_w(Wq, d0 + lane16, c0, hi);
        aq = wmma_bf16(wq_t, bh, aq);       // q from hsv
        v16bf wk_t = load_w(Wk, d0 + lane16, c0, hi);
        ak = wmma_bf16(wk_t, br, ak);       // k from rgb
        v16bf wv_t = load_w(Wv, d0 + lane16, c0, hi);
        av = wmma_bf16(wv_t, br, av);       // v from rgb
      }
      const int t = t0 + nt * 16 + lane16;  // pixel index (D-layout N = lane16)
      v8bf qs, ks;
#pragma unroll
      for (int r = 0; r < 8; ++r) {
        const float bqs = (r < 4) ? bqv0[r] : bqv1[r - 4];
        const float bks = (r < 4) ? bkv0[r] : bkv1[r - 4];
        qs[r] = (__bf16)((aq[r] + bqs) * 0.0625f);   // fold 1/sqrt(256) into q
        ks[r] = (__bf16)(ak[r] + bks);
      }
      *(v8bf*)(qT + ((size_t)b * NPIX + t) * CH + d0 + hi * 8) = qs;
      *(v8bf*)(kT + ((size_t)b * NPIX + t) * CH + d0 + hi * 8) = ks;
#pragma unroll
      for (int r = 0; r < 8; ++r) {
        const float bvs = (r < 4) ? bvv0[r] : bvv1[r - 4];
        vW[((size_t)b * CH + d0 + hi * 8 + r) * NPIX + t] = (__bf16)(av[r] + bvs);
      }
    }
  }
}

// ---------------------------------------------------------------------------
// Stage 2: flash attention. Grid: (NPIX/64, BATCH), 128 threads (4 waves),
// each wave owns a 16-query tile; 32 keys per step. K/V tiles DMA'd into
// double-buffered LDS by the TDM (wave 0 drives it), overlap via TENSORcnt.
// ---------------------------------------------------------------------------
__global__ __launch_bounds__(128, 1) void flash_attn_kernel(
    const __bf16* __restrict__ qT, const __bf16* __restrict__ kT,
    const __bf16* __restrict__ vW, float* __restrict__ out)
{
  constexpr int KP = 264;  // k tile pitch: 256 + 4 dwords pad / 128 dwords
  constexpr int VP = 40;   // v tile pitch:  32 + 4 dwords pad /  16 dwords
  constexpr int PP = 40;   // p tile pitch
  __shared__ __bf16 k_lds[2][32][KP];     // 33.0 KB  [j][c], double buffered
  __shared__ __bf16 v_lds[2][CH][VP];     // 40.0 KB  [c][j], double buffered
  __shared__ __bf16 p_lds[4][16][PP];     //  5.0 KB
  __shared__ float  l_lds[4][16];

  const int b    = blockIdx.y;
  const int iblk = blockIdx.x * 64;
  const int tid  = threadIdx.x;
  const int w = tid >> 5, lane = tid & 31;
  const int lane16 = lane & 15, hi = lane >> 4;
  const int i0 = iblk + w * 16;

  // Output accumulator O^T tiles: M=i (16 queries), N=c (16 x 16 channels).
  v8f acc[16];
  const v8f zero8 = {0.f,0.f,0.f,0.f,0.f,0.f,0.f,0.f};
#pragma unroll
  for (int t = 0; t < 16; ++t) acc[t] = zero8;
  float m_i = -1e30f, l_i = 0.f;

  // Q B-operand rows live in WGP$/L2 (8 KB per wave) -> read from global.
  const __bf16* qbase = qT + ((size_t)b * NPIX + i0 + lane16) * CH + hi * 16;
#pragma unroll
  for (int kc = 0; kc < 8; ++kc)
    __builtin_prefetch(qbase + kc * 32, 0, 1);         // global_prefetch_b8

  const __bf16* kbat = kT + (size_t)b * NPIX * CH;     // [n][c]
  const __bf16* vbat = vW + (size_t)b * CH * NPIX;     // [c][n]
  constexpr int NT = NPIX / 32;                        // 128 key tiles

  // TDM issue for key-tile jt into buffer buf.
  auto issue_tile = [&](int jt, int buf) {
    // K: 32 rows of 256 bf16, rows contiguous (stride 256), pad 4dw/128dw.
    tdm_load_2d((unsigned)(size_t)&k_lds[buf][0][0],
                kbat + (size_t)(jt * 32) * CH,
                /*tile_x=*/256, /*tile_y=*/32,
                /*tensor_x=*/256, /*tensor_y=*/NPIX,
                /*stride=*/256, /*pad_int=*/6, /*pad_amt=*/3);
    // V: 256 rows of 32 bf16 at row stride 4096, pad 4dw/16dw.
    tdm_load_2d((unsigned)(size_t)&v_lds[buf][0][0],
                vbat + jt * 32,
                /*tile_x=*/32, /*tile_y=*/256,
                /*tensor_x=*/NPIX, /*tensor_y=*/256,
                /*stride=*/NPIX, /*pad_int=*/3, /*pad_amt=*/3);
  };

  if (w == 0) issue_tile(0, 0);                        // prologue prefetch

  for (int jt = 0; jt < NT; ++jt) {
    const int cur = jt & 1;
    if (w == 0) {
      if (jt + 1 < NT) {
        issue_tile(jt + 1, cur ^ 1);                   // prefetch next tile
        __builtin_amdgcn_s_wait_tensorcnt(2);          // current tile landed,
      } else {                                         // next 2 still in flight
        __builtin_amdgcn_s_wait_tensorcnt(0);
      }
    }
    __syncthreads();                                   // publish k/v tile

    // S^T[j, i] = sum_c k[c,j] * q_scaled[c,i]; two 16-j subtiles.
    v8f s0 = zero8, s1 = zero8;
#pragma unroll
    for (int kc = 0; kc < 8; ++kc) {
      const int c0 = kc * 32;
      v16bf qb = load16(qbase + c0);
      v16bf ka0 = load16(&k_lds[cur][lane16][c0 + hi * 16]);
      s0 = wmma_bf16(ka0, qb, s0);
      v16bf ka1 = load16(&k_lds[cur][16 + lane16][c0 + hi * 16]);
      s1 = wmma_bf16(ka1, qb, s1);
    }

    // Online softmax; per-lane state is for query i = lane16 (both halves).
    float mx = s0[0];
#pragma unroll
    for (int r = 0; r < 8; ++r) { mx = fmaxf(mx, s0[r]); mx = fmaxf(mx, s1[r]); }
    mx = fmaxf(mx, __shfl_xor(mx, 16, 32));
    const float m_new = fmaxf(m_i, mx);
    const float corr  = __expf(m_i - m_new);
    float rowsum = 0.f;
    v8bf p0, p1;
#pragma unroll
    for (int r = 0; r < 8; ++r) {
      const float e0 = __expf(s0[r] - m_new);
      const float e1 = __expf(s1[r] - m_new);
      rowsum += e0 + e1;
      p0[r] = (__bf16)e0;
      p1[r] = (__bf16)e1;
    }
    rowsum += __shfl_xor(rowsum, 16, 32);
    l_i = l_i * corr + rowsum;
    m_i = m_new;

    // P^T -> LDS in row-major [i][j] so it can be reloaded as an A-operand.
    *(v8bf*)&p_lds[w][lane16][hi * 8]      = p0;   // j = 8*hi + r
    *(v8bf*)&p_lds[w][lane16][16 + hi * 8] = p1;   // j = 16 + 8*hi + r
    // Broadcast per-query correction into the O layout (i = r + 8*hi).
    float cl[16];
#pragma unroll
    for (int x = 0; x < 16; ++x) cl[x] = __shfl(corr, x, 32);
    float cr[8];
#pragma unroll
    for (int r = 0; r < 8; ++r) cr[r] = hi ? cl[r + 8] : cl[r];
#pragma unroll
    for (int t = 0; t < 16; ++t) {
#pragma unroll
      for (int r = 0; r < 8; ++r) acc[t][r] *= cr[r];
    }

    // O^T[i, c] += P[i, j] * V^T[j, c] over all 16 channel tiles.
    v16bf ap = loadA(&p_lds[w][lane16][hi * 8]);
#pragma unroll
    for (int ct = 0; ct < 16; ++ct) {
      v16bf bvt = loadA(&v_lds[cur][ct * 16 + lane16][hi * 8]);
      acc[ct] = wmma_bf16(ap, bvt, acc[ct]);
    }
    __syncthreads();             // tile fully consumed before TDM reuses buffer
  }

  // Normalize by l and store out[b][c][i] (8 consecutive i per lane).
  if (lane < 16) l_lds[w][lane] = 1.0f / l_i;    // same-wave LDS RAW in order
  v4f linv0 = *(const v4f*)&l_lds[w][hi * 8];
  v4f linv1 = *(const v4f*)&l_lds[w][hi * 8 + 4];
#pragma unroll
  for (int ct = 0; ct < 16; ++ct) {
    const int c = ct * 16 + lane16;
    float* op = out + ((size_t)b * CH + c) * NPIX + i0 + hi * 8;
    v4f o0, o1;
#pragma unroll
    for (int r = 0; r < 4; ++r) {
      o0[r] = acc[ct][r]     * linv0[r];
      o1[r] = acc[ct][r + 4] * linv1[r];
    }
    *(v4f*)(op)     = o0;
    *(v4f*)(op + 4) = o1;
  }
}

// ---------------------------------------------------------------------------
extern "C" void kernel_launch(void* const* d_in, const int* in_sizes, int n_in,
                              void* d_out, int out_size, void* d_ws, size_t ws_size,
                              hipStream_t stream) {
  const float* rgb = (const float*)d_in[0];
  const float* hsv = (const float*)d_in[1];
  const float* Wq  = (const float*)d_in[2];
  const float* bq  = (const float*)d_in[3];
  const float* Wk  = (const float*)d_in[4];
  const float* bk  = (const float*)d_in[5];
  const float* Wv  = (const float*)d_in[6];
  const float* bv  = (const float*)d_in[7];
  float* out = (float*)d_out;

  // Workspace: three bf16 [8][4096][256]-sized buffers (16 MB each).
  const size_t BCN = (size_t)BATCH * NPIX * CH;
  __bf16* qT = (__bf16*)d_ws;
  __bf16* kT = qT + BCN;
  __bf16* vW = kT + BCN;

  dim3 g1(NPIX / 32, BATCH);
  qkv_project_kernel<<<g1, 256, 0, stream>>>(rgb, hsv, Wq, bq, Wk, bk, Wv, bv,
                                             qT, kT, vW);
  dim3 g2(NPIX / 64, BATCH);
  flash_attn_kernel<<<g2, 128, 0, stream>>>(qT, kT, vW, out);
}
